// QwenAttention_43173011259550
// MI455X (gfx1250) — matrix operand order repaired and checked by reading, not compiled, not run on hardware
//
#include <hip/hip_runtime.h>
#include <hip/hip_bf16.h>

// ---------------------------------------------------------------------------
// Qwen attention block for MI455X (gfx1250, wave32, WMMA).
// All matmuls on v_wmma_f32_16x16x32_bf16, f32 accumulation.
// GEMM: register + LDS double-buffered pipeline (global loads for tile k+1
// in flight while tile k runs on the WMMA pipe; one barrier per k-step).
// Every WMMA fragment is a contiguous per-lane vector load (b128-class).
// ---------------------------------------------------------------------------

#define HIDDEN 2048
#define N_HEADS 16
#define N_KV_HEADS 4
#define HEAD_DIM 128
#define KV_DIM (N_KV_HEADS * HEAD_DIM)
#define BATCH 2
#define SEQ 2048

typedef __bf16 v16bf __attribute__((ext_vector_type(16)));
typedef __bf16 v8bf  __attribute__((ext_vector_type(8)));
typedef float  v8f   __attribute__((ext_vector_type(8)));

__device__ __forceinline__ unsigned short f32_to_bf16(float f) {
  unsigned u = __builtin_bit_cast(unsigned, f);
  unsigned r = u + 0x7FFFu + ((u >> 16) & 1u);   // round-to-nearest-even
  return (unsigned short)(r >> 16);
}

__device__ __forceinline__ v16bf cat8(v8bf lo, v8bf hi) {
  return __builtin_shufflevector(lo, hi, 0, 1, 2, 3, 4, 5, 6, 7,
                                 8, 9, 10, 11, 12, 13, 14, 15);
}

// ---------------------------------------------------------------------------
// fp32 -> bf16 conversion (plain)
// ---------------------------------------------------------------------------
__global__ void cvt_f32_bf16(const float* __restrict__ in,
                             unsigned short* __restrict__ out, int n) {
  int i = blockIdx.x * blockDim.x + threadIdx.x;
  if (i < n) out[i] = f32_to_bf16(in[i]);
}

// ---------------------------------------------------------------------------
// fp32 [K,N] -> bf16 transposed [N,K] (LDS-tiled 32x32), for weight matrices.
// ---------------------------------------------------------------------------
__global__ __launch_bounds__(256) void cvt_t_f32_bf16(
    const float* __restrict__ in, unsigned short* __restrict__ out,
    int K, int N) {
  __shared__ float tile[32][33];
  int k0 = blockIdx.x * 32, n0 = blockIdx.y * 32;
  int tx = threadIdx.x & 31, ty = threadIdx.x >> 5;  // ty 0..7
#pragma unroll
  for (int i = ty; i < 32; i += 8)
    tile[i][tx] = in[(size_t)(k0 + i) * N + n0 + tx];
  __syncthreads();
#pragma unroll
  for (int i = ty; i < 32; i += 8)
    out[(size_t)(n0 + i) * K + k0 + tx] = f32_to_bf16(tile[tx][i]);
}

// ---------------------------------------------------------------------------
// Tiled bf16 WMMA GEMM: C[M,N] = A[M,K] * B[K,N] (+bias), B passed
// PRE-TRANSPOSED as BT[N,K] row-major. 256 threads = 8 waves (2x4), each wave
// computes 2x2 WMMA tiles. Block tile 64x128, K-step 32.
// Software pipeline: regs hold next tile's global data while current tile
// computes out of double-buffered LDS. One __syncthreads per k-step.
// ---------------------------------------------------------------------------
#define GM 64
#define GN 128
#define GK 32

__global__ __launch_bounds__(256) void gemm_bf16(
    const unsigned short* __restrict__ A, const unsigned short* __restrict__ BT,
    const float* __restrict__ bias, float* __restrict__ C,
    int M, int N, int K) {
  __shared__ __align__(32) unsigned short As[2][GM * GK];   // [GM][GK]
  __shared__ __align__(32) unsigned short BsT[2][GN * GK];  // [GN][GK]

  const int tid  = threadIdx.x;
  const int lane = tid & 31;
  const int wave = tid >> 5;
  const int r    = lane & 15;
  const int half = lane >> 4;
  const int wm   = wave >> 2;      // 0..1
  const int wn   = wave & 3;       // 0..3
  const size_t bm = (size_t)blockIdx.x * GM;
  const size_t bn = (size_t)blockIdx.y * GN;

  // Staging assignments (16B chunks). A: 256 chunks; BT: 512 chunks (2/thd).
  const int arow = tid >> 2, acol = (tid & 3) * 8;
  const int b0row = tid >> 2, b0col = (tid & 3) * 8;        // rows 0..63
  const int b1row = 64 + (tid >> 2), b1col = (tid & 3) * 8; // rows 64..127
  const unsigned short* aglob  = A  + (bm + arow)  * (size_t)K + acol;
  const unsigned short* b0glob = BT + (bn + b0row) * (size_t)K + b0col;
  const unsigned short* b1glob = BT + (bn + b1row) * (size_t)K + b1col;

  v8f acc[2][2];
  acc[0][0] = {}; acc[0][1] = {}; acc[1][0] = {}; acc[1][1] = {};

  uint4 ra, rb0, rb1;  // pipeline registers

  auto load_regs = [&](int k0) {
    ra  = *(const uint4*)(aglob + k0);
    rb0 = *(const uint4*)(b0glob + k0);
    rb1 = *(const uint4*)(b1glob + k0);
  };
  auto store_lds = [&](int buf) {
    *(uint4*)&As[buf][arow * GK + acol]    = ra;
    *(uint4*)&BsT[buf][b0row * GK + b0col] = rb0;
    *(uint4*)&BsT[buf][b1row * GK + b1col] = rb1;
  };
  auto compute = [&](int buf) {
    v16bf afrag[2], bfrag[2];
#pragma unroll
    for (int tm = 0; tm < 2; ++tm) {
      const unsigned short* ap =
          &As[buf][(wm * 32 + tm * 16 + r) * GK + half * 8];
      afrag[tm] = cat8(*(const v8bf*)ap, *(const v8bf*)(ap + 16));
    }
#pragma unroll
    for (int tn = 0; tn < 2; ++tn) {
      const unsigned short* bp =
          &BsT[buf][(wn * 32 + tn * 16 + r) * GK + half * 8];
      bfrag[tn] = cat8(*(const v8bf*)bp, *(const v8bf*)(bp + 16));
    }
#pragma unroll
    for (int tm = 0; tm < 2; ++tm)
#pragma unroll
      for (int tn = 0; tn < 2; ++tn)
        acc[tm][tn] = __builtin_amdgcn_wmma_f32_16x16x32_bf16(
            false, afrag[tm], false, bfrag[tn], (short)0, acc[tm][tn],
            false, false);
  };

  // Prologue: tile 0 into LDS buffer 0.
  load_regs(0);
  store_lds(0);
  __syncthreads();

  int cur = 0;
  for (int k0 = GK; k0 < K; k0 += GK) {
    load_regs(k0);        // next tile's global loads in flight during compute
    compute(cur);         // WMMAs on current LDS buffer
    store_lds(cur ^ 1);   // single loadcnt wait here, then ds_store_b128 x3
    __syncthreads();
    cur ^= 1;
  }
  compute(cur);

  // Write back (C layout: VGPR i -> row half*8+i, col r).
#pragma unroll
  for (int tm = 0; tm < 2; ++tm)
#pragma unroll
    for (int tn = 0; tn < 2; ++tn) {
      size_t colg = bn + wn * 32 + tn * 16 + r;
      float badd = bias ? bias[colg] : 0.0f;
#pragma unroll
      for (int i = 0; i < 8; ++i) {
        size_t rowg = bm + wm * 32 + tm * 16 + half * 8 + i;
        C[rowg * (size_t)N + colg] = acc[tm][tn][i] + badd;
      }
    }
}

// ---------------------------------------------------------------------------
// RoPE (repeat-2 layout): out[e] = x[e]*cos(a_e) + rot(x)[e]*sin(a_e),
// a_e = pos * theta^(-(e>>1)/64). Emits bf16 + optional f32 cache.
// ---------------------------------------------------------------------------
__global__ void rope_kernel(const float* __restrict__ X,
                            unsigned short* __restrict__ Xb,
                            float* __restrict__ cache,
                            int nHeads, int total) {
  int idx = blockIdx.x * blockDim.x + threadIdx.x;
  if (idx >= total) return;
  int e = idx & 127;
  int rest = idx >> 7;
  int s = (rest / nHeads) & (SEQ - 1);
  float ang = (float)s * __powf(10000.0f, -(float)(e >> 1) * (1.0f / 64.0f));
  float c = __cosf(ang), sn = __sinf(ang);
  float x = X[idx];
  float other = (e < 64) ? -X[idx + 64] : X[idx - 64];
  float o = x * c + other * sn;
  Xb[idx] = f32_to_bf16(o);
  if (cache) cache[idx] = o;
}

// V: copy to fp32 cache (reference layout) + bf16 TRANSPOSED [b][kvh][d][s]
// so PV B-fragments are contiguous per lane.
__global__ void vprep_kernel(const float* __restrict__ V,
                             unsigned short* __restrict__ VbT,
                             float* __restrict__ cacheV, int n) {
  int i = blockIdx.x * blockDim.x + threadIdx.x;
  if (i >= n) return;
  float v = V[i];
  cacheV[i] = v;
  int d   = i & 127;
  int kvh = (i >> 7) & 3;
  int s   = (i >> 9) & (SEQ - 1);
  int b   = i >> 20;
  VbT[(((size_t)(b * N_KV_HEADS + kvh) * HEAD_DIM) + d) * SEQ + s] =
      f32_to_bf16(v);
}

// ---------------------------------------------------------------------------
// Flash attention. 128 threads = 4 waves; each wave owns one 16-row Q tile of
// one head. KV tiles of 32; scores + PV via WMMA; online softmax in f32.
// P transposed C-layout -> A-layout through a per-wave LDS tile ordered with
// s_wait_dscnt (CDNA5 split counters). Causal trip count per wave.
// ---------------------------------------------------------------------------
__global__ __launch_bounds__(128) void attn_fa(
    const unsigned short* __restrict__ Q, const unsigned short* __restrict__ Kt,
    const unsigned short* __restrict__ VtT,
    unsigned short* __restrict__ O) {
  __shared__ __align__(16) unsigned short Pbuf[4][16 * 32];  // stride 32 halfs

  const int tid  = threadIdx.x;
  const int lane = tid & 31;
  const int w    = tid >> 5;
  const int r    = lane & 15;
  const int half = lane >> 4;
  const int qt   = blockIdx.x * 4 + w;       // q tile (16 rows)
  const int h    = blockIdx.y;
  const int b    = blockIdx.z;
  const int kvh  = h >> 2;                   // GQA: 4 Q heads per KV head
  const float scale = 0.08838834764831845f;  // 1/sqrt(128)

  // Q fragments: 4 chunks of K=32 over head_dim=128 (2x16B loads each).
  v16bf qf[4];
  {
    const unsigned short* qb =
        Q + (((size_t)(b * SEQ + qt * 16 + r)) * N_HEADS + h) * HEAD_DIM;
#pragma unroll
    for (int c = 0; c < 4; ++c) {
      const unsigned short* p = qb + 32 * c + half * 8;
      qf[c] = cat8(*(const v8bf*)p, *(const v8bf*)(p + 16));
    }
  }

  v8f oacc[8];
#pragma unroll
  for (int dt = 0; dt < 8; ++dt) oacc[dt] = {};
  float mrow[8], lrow[8];
#pragma unroll
  for (int i = 0; i < 8; ++i) { mrow[i] = -3.0e38f; lrow[i] = 0.0f; }

  const int nIter = (qt * 16 + 15 + 32) >> 5;  // causal bound

  const unsigned short* vbase =
      VtT + ((size_t)(b * N_KV_HEADS + kvh) * HEAD_DIM) * SEQ;

  for (int it = 0; it < nIter; ++it) {
    const int kv0 = it * 32;

    // ---- scores S[16x32] = Q (16x128) . K^T (128x32): 8 WMMAs ----
    v8f s[2];
#pragma unroll
    for (int n = 0; n < 2; ++n) {
      s[n] = {};
      const unsigned short* kb =
          Kt + (((size_t)(b * SEQ + kv0 + n * 16 + r)) * N_KV_HEADS + kvh) *
                   HEAD_DIM;
#pragma unroll
      for (int c = 0; c < 4; ++c) {
        v16bf bf = cat8(*(const v8bf*)(kb + 32 * c + half * 8), *(const v8bf*)(kb + 32 * c + half * 8 + 16));  // 2x16B reads
        s[n] = __builtin_amdgcn_wmma_f32_16x16x32_bf16(
            false, qf[c], false, bf, (short)0, s[n], false, false);
      }
    }

    // ---- scale + causal mask + online softmax (row=half*8+i, col=r) ----
    float p[2][8];
#pragma unroll
    for (int n = 0; n < 2; ++n)
#pragma unroll
      for (int i = 0; i < 8; ++i) {
        int qr = qt * 16 + half * 8 + i;
        int kvp = kv0 + n * 16 + r;
        float v = s[n][i] * scale;
        p[n][i] = (kvp > qr) ? -1.0e30f : v;
      }
#pragma unroll
    for (int i = 0; i < 8; ++i) {
      float lm = fmaxf(p[0][i], p[1][i]);
#pragma unroll
      for (int m = 1; m < 16; m <<= 1) lm = fmaxf(lm, __shfl_xor(lm, m, 32));
      float mnew = fmaxf(mrow[i], lm);
      float alpha = __expf(mrow[i] - mnew);
      float p0 = __expf(p[0][i] - mnew);
      float p1 = __expf(p[1][i] - mnew);
      float ls = p0 + p1;
#pragma unroll
      for (int m = 1; m < 16; m <<= 1) ls += __shfl_xor(ls, m, 32);
      lrow[i] = lrow[i] * alpha + ls;
      mrow[i] = mnew;
#pragma unroll
      for (int dt = 0; dt < 8; ++dt) oacc[dt][i] *= alpha;
      p[0][i] = p0; p[1][i] = p1;
    }

    // ---- transpose P (C-layout) -> A-layout via per-wave LDS ----
#pragma unroll
    for (int n = 0; n < 2; ++n)
#pragma unroll
      for (int i = 0; i < 8; ++i)
        Pbuf[w][(half * 8 + i) * 32 + n * 16 + r] = f32_to_bf16(p[n][i]);
    asm volatile("s_wait_dscnt 0" ::: "memory");  // in-wave LDS RAW fence

    const unsigned short* pp = &Pbuf[w][r * 32 + half * 8];
    v16bf pf = cat8(*(const v8bf*)pp, *(const v8bf*)(pp + 16));

    // ---- O += P (16x32) . V (32x128): 8 WMMAs; V transposed so each
    // B frag is two 8-wide s-runs (s = 8*half.., 16+8*half..). ----
#pragma unroll
    for (int dt = 0; dt < 8; ++dt) {
      const unsigned short* vp =
          vbase + (size_t)(dt * 16 + r) * SEQ + kv0 + half * 8;
      v16bf vf = cat8(*(const v8bf*)vp, *(const v8bf*)(vp + 16));
      oacc[dt] = __builtin_amdgcn_wmma_f32_16x16x32_bf16(
          false, pf, false, vf, (short)0, oacc[dt], false, false);
    }
  }

  // ---- normalize + store bf16 attn_out [B,S,H,D] ----
#pragma unroll
  for (int dt = 0; dt < 8; ++dt)
#pragma unroll
    for (int i = 0; i < 8; ++i) {
      int qr = qt * 16 + half * 8 + i;
      float v = oacc[dt][i] / lrow[i];
      O[(((size_t)(b * SEQ + qr)) * N_HEADS + h) * HEAD_DIM + dt * 16 + r] =
          f32_to_bf16(v);
    }
}

// ---------------------------------------------------------------------------
// Host-side launch
// ---------------------------------------------------------------------------
extern "C" void kernel_launch(void* const* d_in, const int* in_sizes, int n_in,
                              void* d_out, int out_size, void* d_ws,
                              size_t ws_size, hipStream_t stream) {
  (void)in_sizes; (void)n_in; (void)out_size; (void)ws_size;

  const float* hs = (const float*)d_in[0];
  // d_in[1] attention_mask (pure causal -> computed in-kernel), d_in[2] pos ids
  const float* wq = (const float*)d_in[3];
  const float* bq = (const float*)d_in[4];
  const float* wk = (const float*)d_in[5];
  const float* bk = (const float*)d_in[6];
  const float* wv = (const float*)d_in[7];
  const float* bv = (const float*)d_in[8];
  const float* wo = (const float*)d_in[9];

  float* out     = (float*)d_out;                           // [B,S,HIDDEN]
  float* cache_k = out + (size_t)BATCH * SEQ * HIDDEN;      // [B,S,4,128]
  float* cache_v = cache_k + (size_t)BATCH * SEQ * KV_DIM;  // [B,S,4,128]

  const size_t M = (size_t)BATCH * SEQ;  // 4096 rows
  char* ws = (char*)d_ws;
  size_t off = 0;
  auto carve = [&](size_t bytes) -> char* {
    char* p = ws + off;
    off += (bytes + 255) & ~(size_t)255;
    return p;
  };
  unsigned short* hs_bf  = (unsigned short*)carve(M * HIDDEN * 2);
  unsigned short* wqT_bf = (unsigned short*)carve((size_t)HIDDEN * HIDDEN * 2);
  unsigned short* wkT_bf = (unsigned short*)carve((size_t)HIDDEN * KV_DIM * 2);
  unsigned short* wvT_bf = (unsigned short*)carve((size_t)HIDDEN * KV_DIM * 2);
  unsigned short* woT_bf = (unsigned short*)carve((size_t)HIDDEN * HIDDEN * 2);
  float*          q_f    = (float*)carve(M * HIDDEN * 4);
  float*          k_f    = (float*)carve(M * KV_DIM * 4);
  float*          v_f    = (float*)carve(M * KV_DIM * 4);
  unsigned short* q_bf   = (unsigned short*)carve(M * HIDDEN * 2);
  unsigned short* k_bf   = (unsigned short*)carve(M * KV_DIM * 2);
  unsigned short* vT_bf  = (unsigned short*)carve(M * KV_DIM * 2);
  unsigned short* ao_bf  = (unsigned short*)carve(M * HIDDEN * 2);

  // 1) activations: plain convert; weights: convert + transpose to [N][K]
  cvt_f32_bf16<<<dim3((unsigned)((M * HIDDEN + 255) / 256)), dim3(256), 0,
                 stream>>>(hs, hs_bf, (int)(M * HIDDEN));
  cvt_t_f32_bf16<<<dim3(HIDDEN / 32, HIDDEN / 32), dim3(256), 0, stream>>>(
      wq, wqT_bf, HIDDEN, HIDDEN);
  cvt_t_f32_bf16<<<dim3(HIDDEN / 32, KV_DIM / 32), dim3(256), 0, stream>>>(
      wk, wkT_bf, HIDDEN, KV_DIM);
  cvt_t_f32_bf16<<<dim3(HIDDEN / 32, KV_DIM / 32), dim3(256), 0, stream>>>(
      wv, wvT_bf, HIDDEN, KV_DIM);
  cvt_t_f32_bf16<<<dim3(HIDDEN / 32, HIDDEN / 32), dim3(256), 0, stream>>>(
      wo, woT_bf, HIDDEN, HIDDEN);

  // 2) projections (WMMA GEMMs; B passed transposed)
  gemm_bf16<<<dim3((unsigned)(M / GM), HIDDEN / GN), dim3(256), 0, stream>>>(
      hs_bf, wqT_bf, bq, q_f, (int)M, HIDDEN, HIDDEN);
  gemm_bf16<<<dim3((unsigned)(M / GM), KV_DIM / GN), dim3(256), 0, stream>>>(
      hs_bf, wkT_bf, bk, k_f, (int)M, KV_DIM, HIDDEN);
  gemm_bf16<<<dim3((unsigned)(M / GM), KV_DIM / GN), dim3(256), 0, stream>>>(
      hs_bf, wvT_bf, bv, v_f, (int)M, KV_DIM, HIDDEN);

  // 3) RoPE (Q -> bf16; K -> bf16 + fp32 cache_k), V prep (bf16^T + cache_v)
  {
    int nq = (int)(M * HIDDEN);
    rope_kernel<<<dim3((nq + 255) / 256), dim3(256), 0, stream>>>(
        q_f, q_bf, nullptr, N_HEADS, nq);
    int nk = (int)(M * KV_DIM);
    rope_kernel<<<dim3((nk + 255) / 256), dim3(256), 0, stream>>>(
        k_f, k_bf, cache_k, N_KV_HEADS, nk);
    vprep_kernel<<<dim3((nk + 255) / 256), dim3(256), 0, stream>>>(
        v_f, vT_bf, cache_v, nk);
  }

  // 4) flash attention (grid: q-tiles/4 x heads x batch)
  attn_fa<<<dim3(SEQ / 64, N_HEADS, BATCH), dim3(128), 0, stream>>>(
      q_bf, k_bf, vT_bf, ao_bf);

  // 5) output projection
  gemm_bf16<<<dim3((unsigned)(M / GM), HIDDEN / GN), dim3(256), 0, stream>>>(
      ao_bf, woT_bf, nullptr, out, (int)M, HIDDEN, HIDDEN);
}
